// selfattention_58591943852126
// MI455X (gfx1250) — matrix operand order repaired and checked by reading, not compiled, not run on hardware
//
#include <hip/hip_runtime.h>
#include <hip/hip_bf16.h>

typedef __attribute__((ext_vector_type(16))) __bf16 bf16x16;
typedef __attribute__((ext_vector_type(8)))  __bf16 bf16x8;
typedef __attribute__((ext_vector_type(2)))  __bf16 bf16x2;
typedef __attribute__((ext_vector_type(8)))  float  f32x8;
typedef __attribute__((ext_vector_type(8)))  float  v8f;
typedef int i32x4 __attribute__((vector_size(16)));
typedef __bf16 bf16_t;

#define AS1 __attribute__((address_space(1)))
#define AS3 __attribute__((address_space(3)))

union AFrag { bf16x16 v; bf16x8 h[2]; };
union BFrag { bf16x16 v; bf16x8 h[2]; };

// Packed f32->bf16 convert: hardware v_cvt_pk_bf16_f32.
static __device__ inline bf16x2 cvt2(float a, float b) {
#if __has_builtin(__builtin_amdgcn_cvt_pk_bf16_f32)
    return __builtin_amdgcn_cvt_pk_bf16_f32(a, b);
#else
    bf16x2 r; r[0] = (bf16_t)a; r[1] = (bf16_t)b; return r;
#endif
}

static __device__ inline bf16_t f2bf(float f) {
    return cvt2(f, 0.0f)[0];
}

static __device__ inline bf16x8 cvt8(f32x8 f) {
    bf16x8 r;
#pragma unroll
    for (int i = 0; i < 4; ++i) {
        bf16x2 p = cvt2(f[2 * i], f[2 * i + 1]);
        r[2 * i] = p[0];
        r[2 * i + 1] = p[1];
    }
    return r;
}

static __device__ inline v8f wmma_bf16(const AFrag& a, const BFrag& b, v8f c) {
    return __builtin_amdgcn_wmma_f32_16x16x32_bf16(
        /*neg_a=*/false, a.v, /*neg_b=*/false, b.v,
        /*c_mod=*/(short)0, c, /*reuse_a=*/false, /*reuse_b=*/false);
}

// Async global->LDS 16-byte copy (ASYNCcnt-tracked, gfx1250-specific path).
// Probe-confirmed: builtin takes (v4i AS1*, v4i AS3*, imm offset, imm cpol).
static __device__ inline void async_copy_b128(void* lds_dst, const void* gsrc) {
#if __has_builtin(__builtin_amdgcn_global_load_async_to_lds_b128)
    __builtin_amdgcn_global_load_async_to_lds_b128(
        (AS1 i32x4*)(void*)gsrc, (AS3 i32x4*)lds_dst, 0, 0);
#else
    unsigned ldsaddr = (unsigned)(unsigned long long)(AS3 void*)lds_dst;
    asm volatile("global_load_async_to_lds_b128 %0, %1, off"
                 :: "v"(ldsaddr), "v"((unsigned long long)gsrc)
                 : "memory");
#endif
}

static __device__ inline void wait_async0() {
#if __has_builtin(__builtin_amdgcn_s_wait_asynccnt)
    __builtin_amdgcn_s_wait_asynccnt(0);
#else
    asm volatile("s_wait_asynccnt 0x0" ::: "memory");
#endif
}

// ---------------------------------------------------------------------------
// Pass 0: elementwise fp32 -> bf16 (bandwidth bound, done once).
// ---------------------------------------------------------------------------
__global__ __launch_bounds__(256) void cvt_bf16_kernel(
    const float* __restrict__ src, bf16_t* __restrict__ dst, int n8)
{
    int i = blockIdx.x * blockDim.x + threadIdx.x;
    if (i < n8) {
        f32x8 v = *(const f32x8*)(src + (size_t)i * 8);
        *(bf16x8*)(dst + (size_t)i * 8) = cvt8(v);
    }
}

// ---------------------------------------------------------------------------
// Pass 1: projection O[m][n] = sum_k X[m][k] * W[n][k], all bf16 in, bf16 out.
// Wave tile 32x32 (2x2 WMMA accs), WG = 8 waves (4x2) -> 128x64 block tile.
// ---------------------------------------------------------------------------
__global__ __launch_bounds__(256) void proj_gemm_kernel(
    const bf16_t* __restrict__ Xb, const bf16_t* __restrict__ Wb,
    bf16_t* __restrict__ P)
{
    const int z = blockIdx.z;
    const bf16_t* __restrict__ X = Xb + (size_t)z * 4096u * 1024u;
    const bf16_t* __restrict__ W = Wb + (size_t)z * 1024u * 1024u;
    bf16_t* __restrict__ O = P + (size_t)z * 4096u * 1024u;

    const int tid  = threadIdx.x;
    const int wave = tid >> 5;
    const int lane = tid & 31;
    const int lo   = lane & 15;
    const int hi   = lane >> 4;
    const int wr   = wave >> 1;      // 0..3
    const int wc   = wave & 1;       // 0..1
    const int m0   = blockIdx.x * 128 + wr * 32;
    const int n0   = blockIdx.y * 64  + wc * 32;

    v8f c00 = {}, c01 = {}, c10 = {}, c11 = {};

    const bf16_t* xr0 = X + (size_t)(m0 + lo) * 1024;
    const bf16_t* xr1 = xr0 + (size_t)16 * 1024;
    const bf16_t* wr0 = W + (size_t)(n0 + lo) * 1024;
    const bf16_t* wr1 = wr0 + (size_t)16 * 1024;

    for (int k0 = 0; k0 < 1024; k0 += 32) {
        AFrag a0, a1;
        a0.h[0] = *(const bf16x8*)(xr0 + k0 + 16 * hi);
        a0.h[1] = *(const bf16x8*)(xr0 + k0 + 16 * hi + 8);
        a1.h[0] = *(const bf16x8*)(xr1 + k0 + 16 * hi);
        a1.h[1] = *(const bf16x8*)(xr1 + k0 + 16 * hi + 8);
        BFrag b0, b1;
        b0.h[0] = *(const bf16x8*)(wr0 + k0 + 16 * hi);
        b0.h[1] = *(const bf16x8*)(wr0 + k0 + 16 * hi + 8);
        b1.h[0] = *(const bf16x8*)(wr1 + k0 + 16 * hi);
        b1.h[1] = *(const bf16x8*)(wr1 + k0 + 16 * hi + 8);

        c00 = wmma_bf16(a0, b0, c00);
        c01 = wmma_bf16(a0, b1, c01);
        c10 = wmma_bf16(a1, b0, c10);
        c11 = wmma_bf16(a1, b1, c11);
    }

#pragma unroll
    for (int r = 0; r < 8; ++r) {
        const int m = r + 8 * hi;
        O[(size_t)(m0 + m)      * 1024 + n0 + lo]      = f2bf(c00[r]);
        O[(size_t)(m0 + m)      * 1024 + n0 + 16 + lo] = f2bf(c01[r]);
        O[(size_t)(m0 + 16 + m) * 1024 + n0 + lo]      = f2bf(c10[r]);
        O[(size_t)(m0 + 16 + m) * 1024 + n0 + 16 + lo] = f2bf(c11[r]);
    }
}

// ---------------------------------------------------------------------------
// Pass 2: flash attention. grid (qblk=16, head=16, n=2), 8 waves; wave owns
// 16 q rows. Per 32-key block: K staged via async global->LDS copy (padded
// stride 72 halves), V staged transposed (stride 40), P reshaped through
// per-wave LDS. All stage strides keep ds_load_b128 16B-aligned and
// bank-conflict-free.
// ---------------------------------------------------------------------------
#define KT_STRIDE 72
#define VT_STRIDE 40
#define P_STRIDE  40

__global__ __launch_bounds__(256) void flash_attn_kernel(
    const bf16_t* __restrict__ vp, const bf16_t* __restrict__ kp,
    const bf16_t* __restrict__ qp, float* __restrict__ out)
{
    constexpr int S = 2048, E = 1024, D = 64;
    __shared__ __align__(16) bf16_t lds[32 * KT_STRIDE + 64 * VT_STRIDE + 8 * 16 * P_STRIDE];
    bf16_t* Kt = lds;
    bf16_t* Vt = lds + 32 * KT_STRIDE;

    const int tid  = threadIdx.x;
    const int wave = tid >> 5;
    const int lane = tid & 31;
    const int lo   = lane & 15;
    const int hi   = lane >> 4;
    const int n    = blockIdx.z;
    const int head = blockIdx.y;
    const int q0   = blockIdx.x * 128 + wave * 16;
    const size_t hb = (size_t)n * S * E + (size_t)head * D;

    // Q fragments (16 rows x 64 d = two 16x32 A frags), resident whole kernel
    AFrag aq0, aq1;
    {
        const bf16_t* qrow = qp + hb + (size_t)(q0 + lo) * E;
        aq0.h[0] = *(const bf16x8*)(qrow + 16 * hi);
        aq0.h[1] = *(const bf16x8*)(qrow + 16 * hi + 8);
        aq1.h[0] = *(const bf16x8*)(qrow + 32 + 16 * hi);
        aq1.h[1] = *(const bf16x8*)(qrow + 40 + 16 * hi);
    }

    float m_run[8], l_run[8];
    v8f o0 = {}, o1 = {}, o2 = {}, o3 = {};
#pragma unroll
    for (int r = 0; r < 8; ++r) { m_run[r] = -1e30f; l_run[r] = 0.0f; }

    bf16_t* Pw = lds + 32 * KT_STRIDE + 64 * VT_STRIDE + wave * 16 * P_STRIDE;

    const int keyl = tid >> 3;          // 0..31
    const int dg   = (tid & 7) * 8;     // 0..56

    for (int k0 = 0; k0 < S; k0 += 32) {
        // ---- stage K (async, untransposed) and V (transposed) tiles ----
        __syncthreads();
        async_copy_b128(Kt + keyl * KT_STRIDE + dg,
                        kp + hb + (size_t)(k0 + keyl) * E + dg);
        {
            bf16x8 vv = *(const bf16x8*)(vp + hb + (size_t)(k0 + keyl) * E + dg);
#pragma unroll
            for (int j = 0; j < 8; ++j) Vt[(dg + j) * VT_STRIDE + keyl] = vv[j];
        }
        wait_async0();
        __syncthreads();

        // ---- QK^T: two 16x16 score frags from LDS K tile ----
        v8f s0 = {}, s1 = {};
        {
            const bf16_t* krow = Kt + lo * KT_STRIDE + 16 * hi;
            BFrag bdl, bdh;
            bdl.h[0] = *(const bf16x8*)(krow);
            bdl.h[1] = *(const bf16x8*)(krow + 8);
            bdh.h[0] = *(const bf16x8*)(krow + 32);
            bdh.h[1] = *(const bf16x8*)(krow + 40);
            s0 = wmma_bf16(aq0, bdl, s0);
            s0 = wmma_bf16(aq1, bdh, s0);
            krow += 16 * KT_STRIDE;
            bdl.h[0] = *(const bf16x8*)(krow);
            bdl.h[1] = *(const bf16x8*)(krow + 8);
            bdh.h[0] = *(const bf16x8*)(krow + 32);
            bdh.h[1] = *(const bf16x8*)(krow + 40);
            s1 = wmma_bf16(aq0, bdl, s1);
            s1 = wmma_bf16(aq1, bdh, s1);
        }

        // ---- online softmax (rows live across 16-lane groups) ----
        float p0[8], p1[8], alpha[8];
#pragma unroll
        for (int r = 0; r < 8; ++r) {
            float a = s0[r] * 0.125f;
            float b = s1[r] * 0.125f;
            float mx = fmaxf(a, b);
            mx = fmaxf(mx, __shfl_xor(mx, 1, 32));
            mx = fmaxf(mx, __shfl_xor(mx, 2, 32));
            mx = fmaxf(mx, __shfl_xor(mx, 4, 32));
            mx = fmaxf(mx, __shfl_xor(mx, 8, 32));
            const float mn = fmaxf(m_run[r], mx);
            const float al = __expf(m_run[r] - mn);
            const float e0 = __expf(a - mn);
            const float e1 = __expf(b - mn);
            float rs = e0 + e1;
            rs += __shfl_xor(rs, 1, 32);
            rs += __shfl_xor(rs, 2, 32);
            rs += __shfl_xor(rs, 4, 32);
            rs += __shfl_xor(rs, 8, 32);
            l_run[r] = l_run[r] * al + rs;
            m_run[r] = mn;
            alpha[r] = al;
            p0[r] = e0;
            p1[r] = e1;
        }
#pragma unroll
        for (int r = 0; r < 8; ++r) {
            o0[r] *= alpha[r]; o1[r] *= alpha[r];
            o2[r] *= alpha[r]; o3[r] *= alpha[r];
        }

        // ---- reshape P (C layout -> A layout) through per-wave LDS ----
#pragma unroll
        for (int r = 0; r < 8; ++r) {
            const int m = r + 8 * hi;
            bf16x2 pk = cvt2(p0[r], p1[r]);
            Pw[m * P_STRIDE + lo]      = pk[0];
            Pw[m * P_STRIDE + 16 + lo] = pk[1];
        }
        AFrag ap;  // intra-wave DS ordering: no barrier needed
        {
            const bf16_t* prow = Pw + lo * P_STRIDE;
            ap.h[0] = *(const bf16x8*)(prow + 16 * hi);
            ap.h[1] = *(const bf16x8*)(prow + 16 * hi + 8);
        }

        // ---- O += P @ V  (B frags are contiguous rows of Vt) ----
        {
            BFrag bv;
            const bf16_t* vt0 = Vt + (size_t)lo * VT_STRIDE + 16 * hi;
            bv.h[0] = *(const bf16x8*)(vt0);
            bv.h[1] = *(const bf16x8*)(vt0 + 8);
            o0 = wmma_bf16(ap, bv, o0);
            const bf16_t* vt1 = vt0 + 16 * VT_STRIDE;
            bv.h[0] = *(const bf16x8*)(vt1);
            bv.h[1] = *(const bf16x8*)(vt1 + 8);
            o1 = wmma_bf16(ap, bv, o1);
            const bf16_t* vt2 = vt1 + 16 * VT_STRIDE;
            bv.h[0] = *(const bf16x8*)(vt2);
            bv.h[1] = *(const bf16x8*)(vt2 + 8);
            o2 = wmma_bf16(ap, bv, o2);
            const bf16_t* vt3 = vt2 + 16 * VT_STRIDE;
            bv.h[0] = *(const bf16x8*)(vt3);
            bv.h[1] = *(const bf16x8*)(vt3 + 8);
            o3 = wmma_bf16(ap, bv, o3);
        }
    }

    // ---- finalize: divide by l, store f32 output [n, q, head*64 + d] ----
#pragma unroll
    for (int r = 0; r < 8; ++r) {
        const float inv = 1.0f / l_run[r];
        const int m = r + 8 * hi;
        const size_t row = ((size_t)n * S + q0 + m) * E + (size_t)head * D;
        out[row + lo]      = o0[r] * inv;
        out[row + 16 + lo] = o1[r] * inv;
        out[row + 32 + lo] = o2[r] * inv;
        out[row + 48 + lo] = o3[r] * inv;
    }
}

extern "C" void kernel_launch(void* const* d_in, const int* in_sizes, int n_in,
                              void* d_out, int out_size, void* d_ws, size_t ws_size,
                              hipStream_t stream) {
    const float* values  = (const float*)d_in[0];
    const float* keys    = (const float*)d_in[1];
    const float* queries = (const float*)d_in[2];
    const float* Wv      = (const float*)d_in[3];
    const float* Wk      = (const float*)d_in[4];
    const float* Wq      = (const float*)d_in[5];
    float* out = (float*)d_out;

    const size_t NE = (size_t)4096 * 1024;   // per X / per projection
    const size_t WE = (size_t)1024 * 1024;   // per weight matrix

    // workspace layout (bf16 halves): Xb[3*NE] | Wb[3*WE] | P[3*NE]
    bf16_t* xb = (bf16_t*)d_ws;
    bf16_t* wb = xb + 3 * NE;
    bf16_t* pb = wb + 3 * WE;
    bf16_t* vp = pb;            // slot 0
    bf16_t* kp = pb + NE;       // slot 1
    bf16_t* qp = pb + 2 * NE;   // slot 2

    // Pass 0: convert inputs and weights to bf16 (once).
    {
        const int n8x = (int)(NE / 8);
        const int n8w = (int)(WE / 8);
        const int blk = 256;
        cvt_bf16_kernel<<<(n8x + blk - 1) / blk, blk, 0, stream>>>(values,  xb,           n8x);
        cvt_bf16_kernel<<<(n8x + blk - 1) / blk, blk, 0, stream>>>(keys,    xb + NE,      n8x);
        cvt_bf16_kernel<<<(n8x + blk - 1) / blk, blk, 0, stream>>>(queries, xb + 2 * NE,  n8x);
        cvt_bf16_kernel<<<(n8w + blk - 1) / blk, blk, 0, stream>>>(Wv,      wb,           n8w);
        cvt_bf16_kernel<<<(n8w + blk - 1) / blk, blk, 0, stream>>>(Wk,      wb + WE,      n8w);
        cvt_bf16_kernel<<<(n8w + blk - 1) / blk, blk, 0, stream>>>(Wq,      wb + 2 * WE,  n8w);
    }

    // Pass 1: three projections (z = 0,1,2 -> v,k,q slots).
    dim3 gemm_grid(32, 16, 3);  // 4096/128, 1024/64, {v,k,q}
    proj_gemm_kernel<<<gemm_grid, 256, 0, stream>>>(xb, wb, pb);

    // Pass 2: attention.
    dim3 attn_grid(16, 16, 2);  // 2048/128 q-blocks, 16 heads, batch 2
    flash_attn_kernel<<<attn_grid, 256, 0, stream>>>(vp, kp, qp, out);

    (void)in_sizes; (void)n_in; (void)out_size; (void)ws_size;
}
